// DGCNModule_58042188038334
// MI455X (gfx1250) — matrix operand order repaired and checked by reading, not compiled, not run on hardware
//
#include <hip/hip_runtime.h>
#include <hip/hip_bf16.h>
#include <cstddef>
#include <cstdint>

// ---------------- CDNA5 WMMA bf16 helpers (gfx1250, wave32) ----------------
typedef __bf16 bf16_t;
typedef __attribute__((ext_vector_type(16))) __bf16 v16bf;
typedef __attribute__((ext_vector_type(8)))  __bf16 v8bf;
typedef __attribute__((ext_vector_type(8)))  float  v8f;
typedef __attribute__((ext_vector_type(4)))  unsigned int v4u_t;
typedef __attribute__((ext_vector_type(8)))  int v8i_t;
typedef __attribute__((ext_vector_type(4)))  int v4i_t;

#if defined(__has_builtin)
#if __has_builtin(__builtin_amdgcn_tensor_load_to_lds)
#define HAVE_TDM 1
#endif
#endif

__device__ __forceinline__ v8f wmma_bf16(v16bf a, v16bf b, v8f c) {
  // D = A(16x32 bf16) x B(32x16 bf16) + C(16x16 f32)
  return __builtin_amdgcn_wmma_f32_16x16x32_bf16(
      /*neg_a=*/false, a, /*neg_b=*/false, b,
      /*c_mod=*/(short)0, c, /*reuse_a=*/false, /*reuse_b=*/false);
}

// A fragment: row-major A[M,Kp] (Kp multiple of 32). lane l -> row m0+(l&15),
// k = k0 + 16*(l>>4) + e.
__device__ __forceinline__ v16bf load_a_frag(const bf16_t* A, int lda, int m0,
                                             int k0, int Mclamp) {
  int l = threadIdx.x & 31;
  int m = m0 + (l & 15);
  if (m >= Mclamp) m = Mclamp - 1;
  const bf16_t* p = A + (size_t)m * lda + k0 + 16 * (l >> 4);
  v8bf lo = *(const v8bf*)p;
  v8bf hi = *(const v8bf*)(p + 8);
  v16bf r;
#pragma unroll
  for (int i = 0; i < 8; ++i) { r[i] = lo[i]; r[i + 8] = hi[i]; }
  return r;
}

// B fragment from Bt[N,Kp] in global memory (used by the GRU scan kernel):
// lane l -> column n0+(l&15), k = k0 + 16*(l>>4) + e  (32B contiguous load).
__device__ __forceinline__ v16bf load_b_frag(const bf16_t* Bt, int ldb, int n0,
                                             int k0, int Nclamp) {
  int l = threadIdx.x & 31;
  int n = n0 + (l & 15);
  if (n >= Nclamp) n = Nclamp - 1;
  const bf16_t* p = Bt + (size_t)n * ldb + k0 + 16 * (l >> 4);
  return *(const v16bf*)p;
}

// B fragment from an LDS panel holding 16 rows of Kp halves (row-major).
__device__ __forceinline__ v16bf load_b_frag_lds(const bf16_t* panel, int Kp,
                                                 int k0) {
  int l = threadIdx.x & 31;
  const bf16_t* p = panel + (size_t)(l & 15) * Kp + k0 + 16 * (l >> 4);
  return *(const v16bf*)p;  // -> ds_load_b128 x2
}

// ---------------- Generic WMMA GEMM: C[M,Nn] = A[M,Kp]*Bt[Nn,Kp]^T (+bias,relu)
// Requirement: ldb == Kp (all packed weight buffers satisfy this), so the
// block's B panel (rows n0..n0+15) is one contiguous 16*Kp*2-byte region that
// the Tensor Data Mover stages into LDS once per block.
__global__ __launch_bounds__(256) void gemm_bf16_wmma(
    const bf16_t* __restrict__ A, int lda, const bf16_t* __restrict__ Bt,
    int ldb, const float* __restrict__ bias, float* __restrict__ C, int ldc,
    int M, int Nn, int Kp, int relu) {
  __shared__ bf16_t bpanel[16 * 320];  // <= 10 KB
  int wave = threadIdx.x >> 5;
  int l = threadIdx.x & 31;
  int m0 = (blockIdx.y * 8 + wave) * 16;
  if (m0 >= M) m0 = (M > 16) ? (M - 16) : 0;  // keep barriers uniform
  int n0 = blockIdx.x * 16;
  int rows_valid = Nn - n0;
  if (rows_valid > 16) rows_valid = 16;

#ifdef HAVE_TDM
  // --- stage B panel into LDS via the Tensor Data Mover ---
  if (wave == 0) {
    typedef __attribute__((address_space(3))) bf16_t* lds_bf16_p;
    unsigned lds_addr = (unsigned)(size_t)(lds_bf16_p)(void*)bpanel;
    unsigned long long ga =
        (unsigned long long)(const void*)(Bt + (size_t)n0 * ldb);
    unsigned tile_elems = (unsigned)(16 * Kp);          // bf16 elements
    unsigned valid_elems = (unsigned)(rows_valid * Kp); // OOB tail -> zeros
    v4u_t g0 = {1u,                                     // count=1 (valid D#)
                lds_addr,                               // lds_addr
                (unsigned)ga,                           // global_addr[31:0]
                (unsigned)((ga >> 32) & 0x1ffffffull) | 0x80000000u};  // type=2
    v8i_t g1 = {(int)(1u << 16),                             // data_size=2B
                (int)((valid_elems & 0xffffu) << 16),        // tensor_dim0 lo
                (int)(((valid_elems >> 16) & 0xffffu) | (1u << 16)), // dim0 hi | tensor_dim1=1
                (int)((tile_elems & 0xffffu) << 16),         // tile_dim0
                1,                                           // tile_dim1=1
                (int)tile_elems,                             // dim0_stride lo
                0, 0};
    v4i_t gz = {0, 0, 0, 0};
#if __clang_major__ >= 23
    v8i_t gz8 = {0, 0, 0, 0, 0, 0, 0, 0};
    __builtin_amdgcn_tensor_load_to_lds(g0, g1, gz, gz, gz8, 0);
#else
    __builtin_amdgcn_tensor_load_to_lds(g0, g1, gz, gz, 0);
#endif
    __builtin_amdgcn_s_wait_tensorcnt(0);
  }
  __syncthreads();
#else
  // --- fallback: cooperative synchronous staging ---
  {
    int cpr = Kp >> 3;  // 16-byte chunks per row
    for (int t = threadIdx.x; t < 16 * cpr; t += 256) {
      int r = t / cpr, c = t - r * cpr;
      v8bf v;
      if (r < rows_valid) {
        v = *(const v8bf*)(Bt + ((size_t)(n0 + r)) * ldb + c * 8);
      } else {
#pragma unroll
        for (int i = 0; i < 8; ++i) v[i] = (bf16_t)0.0f;
      }
      *(v8bf*)&bpanel[(size_t)r * Kp + c * 8] = v;
    }
  }
  __syncthreads();
#endif

  v8f acc = {};
  for (int k = 0; k < Kp; k += 32) {
    if (k + 32 < Kp)  // global_prefetch_b8 of next A strip
      __builtin_prefetch(A + (size_t)m0 * lda + k + 32, 0, 1);
    v16bf a = load_a_frag(A, lda, m0, k, M);
    v16bf b = load_b_frag_lds(bpanel, Kp, k);
    acc = wmma_bf16(a, b, acc);
  }
  int col = n0 + (l & 15);
  if (col < Nn) {
    float bv = bias ? bias[col] : 0.0f;
#pragma unroll
    for (int r = 0; r < 8; ++r) {
      int row = m0 + r + ((l >> 4) << 3);
      if (row < M) {
        float v = acc[r] + bv;
        if (relu) v = fmaxf(v, 0.0f);
        C[(size_t)row * ldc + col] = v;
      }
    }
  }
}

// ---------------- packing kernels ----------------
// dst[r][k<Kp] = (k<K) ? src[r][k] : 0   (row-major f32 -> padded bf16)
__global__ void pack_rows_bf16(const float* __restrict__ src,
                               bf16_t* __restrict__ dst, int rows, int K,
                               int Kp) {
  int idx = blockIdx.x * blockDim.x + threadIdx.x;
  if (idx >= rows * Kp) return;
  int r = idx / Kp, k = idx - r * Kp;
  dst[idx] = (k < K) ? (bf16_t)src[(size_t)r * K + k] : (bf16_t)0.0f;
}

// dst[o][k<Kp] = (k<K) ? src[k][o] : 0   (transpose f32[K,O] -> bf16[O,Kp])
__global__ void pack_cols_bf16(const float* __restrict__ src,
                               bf16_t* __restrict__ dst, int O, int K, int Kp) {
  int idx = blockIdx.x * blockDim.x + threadIdx.x;
  if (idx >= O * Kp) return;
  int o = idx / Kp, k = idx - o * Kp;
  dst[idx] = (k < K) ? (bf16_t)src[(size_t)k * O + o] : (bf16_t)0.0f;
}

// wrelT[r*100+o][k] = sum_b comp[r,b]*bases[b,k,o]   (k<200, pad to 224)
__global__ void compose_wrel(const float* __restrict__ comp,
                             const float* __restrict__ bases,
                             bf16_t* __restrict__ dst) {
  int idx = blockIdx.x * blockDim.x + threadIdx.x;
  if (idx >= 8 * 100 * 224) return;
  int k = idx % 224;
  int ro = idx / 224;
  int o = ro % 100, r = ro / 100;
  float s = 0.0f;
  if (k < 200) {
    for (int b = 0; b < 30; ++b)
      s += comp[r * 30 + b] * bases[((size_t)b * 200 + k) * 100 + o];
  }
  dst[idx] = (bf16_t)s;
}

// ---------------- persistent bi-GRU scan (1 block per direction) ------------
// gi: [B*L, 600] precomputed x@Wih^T+bih (cols d*300..). whhT: [600,128] bf16.
__global__ __launch_bounds__(320) void gru_scan(
    const float* __restrict__ gi, const bf16_t* __restrict__ whhT,
    const float* __restrict__ bhh, float* __restrict__ hout) {
  __shared__ bf16_t hprev[32][128];
  __shared__ float gh[32][304];
  const int d = blockIdx.x;  // 0 = fwd, 1 = bwd
  const int tid = threadIdx.x;
  const int wave = tid >> 5, l = tid & 31;
  const bf16_t* Bt = whhT + (size_t)d * 300 * 128;
  for (int i = tid; i < 32 * 128; i += 320) ((bf16_t*)hprev)[i] = (bf16_t)0.0f;
  __syncthreads();
  for (int step = 0; step < 512; ++step) {
    int t = d ? (511 - step) : step;
    // gh = hprev @ whh^T : M=32 (2 tiles) x N=300 (19 tiles), K=128
    for (int tile = wave; tile < 38; tile += 10) {
      int m0 = (tile & 1) * 16;
      int n0 = (tile >> 1) * 16;
      v8f acc = {};
      for (int k = 0; k < 128; k += 32) {
        int m = m0 + (l & 15);
        const bf16_t* p = &hprev[m][k + 16 * (l >> 4)];
        v8bf lo = *(const v8bf*)p;          // ds_load_b128
        v8bf hi = *(const v8bf*)(p + 8);
        v16bf a;
#pragma unroll
        for (int i = 0; i < 8; ++i) { a[i] = lo[i]; a[i + 8] = hi[i]; }
        v16bf b = load_b_frag(Bt, 128, n0, k, 300);
        acc = wmma_bf16(a, b, acc);
      }
      int col = n0 + (l & 15);
      if (col < 300) {
#pragma unroll
        for (int r = 0; r < 8; ++r) gh[m0 + r + ((l >> 4) << 3)][col] = acc[r];
      }
    }
    __syncthreads();
    // gate math + state update (each (b,j) owned by exactly one thread)
    for (int i = tid; i < 3200; i += 320) {
      int b = i / 100, j = i - b * 100;
      size_t g = ((size_t)(b * 512 + t)) * 600 + d * 300;
      float ir = gi[g + j], iz = gi[g + 100 + j], in = gi[g + 200 + j];
      float hr = gh[b][j] + bhh[d * 300 + j];
      float hz = gh[b][100 + j] + bhh[d * 300 + 100 + j];
      float hn = gh[b][200 + j] + bhh[d * 300 + 200 + j];
      float hp = (float)hprev[b][j];
      float rg = 1.0f / (1.0f + __expf(-(ir + hr)));
      float zg = 1.0f / (1.0f + __expf(-(iz + hz)));
      float ng = tanhf(in + rg * hn);
      float hv = (1.0f - zg) * ng + zg * hp;
      hout[((size_t)(b * 512 + t)) * 200 + d * 100 + j] = hv;
      hprev[b][j] = (bf16_t)hv;
    }
    __syncthreads();
  }
}

// ---------------- windowed edge attention softmax (1 wave per (b,j)) --------
__global__ __launch_bounds__(32) void edge_att(const float* __restrict__ hW,
                                               const float* __restrict__ h,
                                               float* __restrict__ alpha) {
  int bj = blockIdx.x;          // b*512 + j
  int j = bj & 511, b = bj >> 9;
  int s = threadIdx.x;          // window slot, k = j-10+s
  int k = j - 10 + s;
  bool valid = (s < 21) && (k >= 0) && (k < 512);
  float score = -1e30f;
  if (valid) {
    const float* x = hW + (size_t)bj * 200;
    const float* y = h + ((size_t)(b * 512 + k)) * 200;
    float acc = 0.0f;
    for (int dd = 0; dd < 200; ++dd) acc += x[dd] * y[dd];
    score = acc;
  }
  float m = score;
#pragma unroll
  for (int off = 16; off > 0; off >>= 1) m = fmaxf(m, __shfl_xor(m, off, 32));
  float e = valid ? __expf(score - m) : 0.0f;
  float sum = e;
#pragma unroll
  for (int off = 16; off > 0; off >>= 1) sum += __shfl_xor(sum, off, 32);
  if (s < 21) alpha[(size_t)bj * 21 + s] = valid ? (e / sum) : 0.0f;
}

// ---------------- RGCN messages: agg[dst] += x_rel[etype,src]*norm ----------
__global__ __launch_bounds__(128) void rgcn_msg(
    const float* __restrict__ x_rel, const float* __restrict__ alpha,
    const int* __restrict__ eb, const int* __restrict__ ej,
    const int* __restrict__ ek, const int* __restrict__ spk,
    float* __restrict__ agg, float* __restrict__ cnt, int E) {
  int e = blockIdx.x;
  if (e >= E) return;
  int b = eb[e], j = ej[e], k = ek[e];
  int src = b * 512 + j, dst = b * 512 + k;
  int etype = 2 * (spk[src] * 2 + spk[dst]) + ((j < k) ? 0 : 1);
  float norm = alpha[(size_t)src * 21 + (k - j + 10)];
  int o = threadIdx.x;
  if (o < 100)
    atomicAdd(&agg[(size_t)dst * 100 + o],
              x_rel[(size_t)src * 800 + etype * 100 + o] * norm);
  if (o == 0) atomicAdd(&cnt[dst], 1.0f);
}

__global__ __launch_bounds__(128) void gc_msg(const float* __restrict__ h1,
                                              const int* __restrict__ eb,
                                              const int* __restrict__ ej,
                                              const int* __restrict__ ek,
                                              float* __restrict__ agg2, int E) {
  int e = blockIdx.x;
  if (e >= E) return;
  int src = eb[e] * 512 + ej[e], dst = eb[e] * 512 + ek[e];
  int o = threadIdx.x;
  if (o < 100)
    atomicAdd(&agg2[(size_t)dst * 100 + o], h1[(size_t)src * 100 + o]);
}

// ---------------- elementwise combiners ----------------
__global__ void h1_combine(const float* __restrict__ agg,
                           const float* __restrict__ cnt,
                           const float* __restrict__ roott,
                           const float* __restrict__ bias,
                           float* __restrict__ h1, int N) {
  int idx = blockIdx.x * blockDim.x + threadIdx.x;
  if (idx >= N * 100) return;
  int n = idx / 100, o = idx - n * 100;
  h1[idx] = agg[idx] / fmaxf(cnt[n], 1.0f) + roott[idx] + bias[o];
}

__global__ void h2_combine(const float* __restrict__ t1,
                           const float* __restrict__ t2,
                           const float* __restrict__ bias,
                           float* __restrict__ h2, int N) {
  int idx = blockIdx.x * blockDim.x + threadIdx.x;
  if (idx >= N * 100) return;
  h2[idx] = t1[idx] + t2[idx] + bias[idx % 100];
}

// z = [feats(200) ; h2(100) ; pad] -> bf16 [N,320]
__global__ void pack_z(const float* __restrict__ feats,
                       const float* __restrict__ h2, bf16_t* __restrict__ z,
                       int N) {
  int idx = blockIdx.x * blockDim.x + threadIdx.x;
  if (idx >= N * 320) return;
  int n = idx / 320, k = idx - n * 320;
  float v = 0.0f;
  if (k < 200) v = feats[(size_t)n * 200 + k];
  else if (k < 300) v = h2[(size_t)n * 100 + (k - 200)];
  z[idx] = (bf16_t)v;
}

// logits = relu'd zz @ clf_w2 + b2
__global__ void clf_final(const float* __restrict__ zz,
                          const float* __restrict__ w2,
                          const float* __restrict__ b2,
                          float* __restrict__ logits, int N) {
  int n = blockIdx.x * blockDim.x + threadIdx.x;
  if (n >= N) return;
  float a0 = b2[0], a1 = b2[1], a2 = b2[2], a3 = b2[3];
  const float* row = zz + (size_t)n * 100;
  for (int j = 0; j < 100; ++j) {
    float v = row[j];
    a0 += v * w2[j * 4 + 0];
    a1 += v * w2[j * 4 + 1];
    a2 += v * w2[j * 4 + 2];
    a3 += v * w2[j * 4 + 3];
  }
  logits[(size_t)n * 4 + 0] = a0;
  logits[(size_t)n * 4 + 1] = a1;
  logits[(size_t)n * 4 + 2] = a2;
  logits[(size_t)n * 4 + 3] = a3;
}

// ------------------------------ launcher ------------------------------------
extern "C" void kernel_launch(void* const* d_in, const int* in_sizes, int n_in,
                              void* d_out, int out_size, void* d_ws,
                              size_t ws_size, hipStream_t stream) {
  (void)n_in; (void)out_size; (void)ws_size;
  const int N = 32 * 512;
  const int E = in_sizes[3];

  const float* input = (const float*)d_in[0];
  const int*   spk   = (const int*)d_in[1];
  const int*   eb    = (const int*)d_in[3];
  const int*   ej    = (const int*)d_in[4];
  const int*   ek    = (const int*)d_in[5];
  const float* wih0  = (const float*)d_in[6];
  const float* whh0  = (const float*)d_in[7];
  const float* bih0  = (const float*)d_in[8];
  const float* bhh0  = (const float*)d_in[9];
  const float* wih1  = (const float*)d_in[10];
  const float* whh1  = (const float*)d_in[11];
  const float* bih1  = (const float*)d_in[12];
  const float* bhh1  = (const float*)d_in[13];
  const float* attw  = (const float*)d_in[14];
  const float* bases = (const float*)d_in[15];
  const float* comp  = (const float*)d_in[16];
  const float* root  = (const float*)d_in[17];
  const float* rbias = (const float*)d_in[18];
  const float* gcw   = (const float*)d_in[19];
  const float* gcr   = (const float*)d_in[20];
  const float* gcb   = (const float*)d_in[21];
  const float* cw1   = (const float*)d_in[22];
  const float* cb1   = (const float*)d_in[23];
  const float* cw2   = (const float*)d_in[24];
  const float* cb2   = (const float*)d_in[25];

  float* logits = (float*)d_out;                  // [N,4]
  float* h2     = (float*)d_out + (size_t)N * 4;  // [N,100]

  char* ws = (char*)d_ws;
  size_t off = 0;
  auto carve = [&](size_t bytes) -> char* {
    char* p = ws + off;
    off += (bytes + 255) & ~(size_t)255;
    return p;
  };
  bf16_t* x0_bf   = (bf16_t*)carve((size_t)N * 128 * 2);
  bf16_t* wih0_bf = (bf16_t*)carve(600 * 128 * 2);
  bf16_t* whh0_bf = (bf16_t*)carve(600 * 128 * 2);
  bf16_t* wih1_bf = (bf16_t*)carve(600 * 224 * 2);
  bf16_t* whh1_bf = (bf16_t*)carve(600 * 128 * 2);
  bf16_t* attwT   = (bf16_t*)carve(200 * 224 * 2);
  bf16_t* wrelT   = (bf16_t*)carve(800 * 224 * 2);
  bf16_t* rootT   = (bf16_t*)carve(100 * 224 * 2);
  bf16_t* gcwT    = (bf16_t*)carve(100 * 128 * 2);
  bf16_t* gcrT    = (bf16_t*)carve(100 * 128 * 2);
  bf16_t* cw1T    = (bf16_t*)carve(100 * 320 * 2);
  float*  GI      = (float*)carve((size_t)N * 600 * 4);
  float*  h0      = (float*)carve((size_t)N * 200 * 4);
  bf16_t* h0_bf   = (bf16_t*)carve((size_t)N * 224 * 2);
  float*  feats   = (float*)carve((size_t)N * 200 * 4);
  bf16_t* feats_bf= (bf16_t*)carve((size_t)N * 224 * 2);
  float*  hW      = (float*)carve((size_t)N * 200 * 4);
  float*  alpha   = (float*)carve((size_t)N * 21 * 4);
  float*  x_rel   = (float*)carve((size_t)N * 800 * 4);
  float*  agg     = (float*)carve((size_t)N * 100 * 4);
  float*  cnt     = (float*)carve((size_t)N * 4);
  float*  tmp1    = (float*)carve((size_t)N * 100 * 4);
  float*  h1      = (float*)carve((size_t)N * 100 * 4);
  bf16_t* h1_bf   = (bf16_t*)carve((size_t)N * 128 * 2);
  float*  agg2    = (float*)carve((size_t)N * 100 * 4);
  bf16_t* agg2_bf = (bf16_t*)carve((size_t)N * 128 * 2);
  float*  t1      = (float*)carve((size_t)N * 100 * 4);
  float*  t2      = (float*)carve((size_t)N * 100 * 4);
  bf16_t* z_bf    = (bf16_t*)carve((size_t)N * 320 * 2);
  float*  zz      = (float*)carve((size_t)N * 100 * 4);

  hipMemsetAsync(agg, 0, (size_t)N * 100 * 4, stream);
  hipMemsetAsync(cnt, 0, (size_t)N * 4, stream);
  hipMemsetAsync(agg2, 0, (size_t)N * 100 * 4, stream);

  auto g1 = [](int n) { return (n + 255) / 256; };
  // weight packing
  pack_rows_bf16<<<g1(N * 128), 256, 0, stream>>>(input, x0_bf, N, 100, 128);
  pack_rows_bf16<<<g1(600 * 128), 256, 0, stream>>>(wih0, wih0_bf, 600, 100, 128);
  pack_rows_bf16<<<g1(600 * 128), 256, 0, stream>>>(whh0, whh0_bf, 600, 100, 128);
  pack_rows_bf16<<<g1(600 * 224), 256, 0, stream>>>(wih1, wih1_bf, 600, 200, 224);
  pack_rows_bf16<<<g1(600 * 128), 256, 0, stream>>>(whh1, whh1_bf, 600, 100, 128);
  pack_cols_bf16<<<g1(200 * 224), 256, 0, stream>>>(attw, attwT, 200, 200, 224);
  compose_wrel<<<g1(8 * 100 * 224), 256, 0, stream>>>(comp, bases, wrelT);
  pack_cols_bf16<<<g1(100 * 224), 256, 0, stream>>>(root, rootT, 100, 200, 224);
  pack_cols_bf16<<<g1(100 * 128), 256, 0, stream>>>(gcw, gcwT, 100, 100, 128);
  pack_cols_bf16<<<g1(100 * 128), 256, 0, stream>>>(gcr, gcrT, 100, 100, 128);
  pack_cols_bf16<<<g1(100 * 320), 256, 0, stream>>>(cw1, cw1T, 100, 300, 320);

  const int GY = (N + 127) / 128;
  // --- layer 0: GI = X @ Wih0^T + bih0 (both directions fused, Nn=600) ---
  gemm_bf16_wmma<<<dim3(38, GY), 256, 0, stream>>>(x0_bf, 128, wih0_bf, 128,
                                                   bih0, GI, 600, N, 600, 128, 0);
  gru_scan<<<2, 320, 0, stream>>>(GI, whh0_bf, bhh0, h0);
  pack_rows_bf16<<<g1(N * 224), 256, 0, stream>>>(h0, h0_bf, N, 200, 224);
  // --- layer 1 ---
  gemm_bf16_wmma<<<dim3(38, GY), 256, 0, stream>>>(h0_bf, 224, wih1_bf, 224,
                                                   bih1, GI, 600, N, 600, 224, 0);
  gru_scan<<<2, 320, 0, stream>>>(GI, whh1_bf, bhh1, feats);
  pack_rows_bf16<<<g1(N * 224), 256, 0, stream>>>(feats, feats_bf, N, 200, 224);

  // --- edge attention ---
  gemm_bf16_wmma<<<dim3(13, GY), 256, 0, stream>>>(feats_bf, 224, attwT, 224,
                                                   nullptr, hW, 200, N, 200, 224, 0);
  edge_att<<<N, 32, 0, stream>>>(hW, feats, alpha);

  // --- RGCN: x_rel for all 8 relations in one GEMM (Nn=800) ---
  gemm_bf16_wmma<<<dim3(50, GY), 256, 0, stream>>>(feats_bf, 224, wrelT, 224,
                                                   nullptr, x_rel, 800, N, 800, 224, 0);
  rgcn_msg<<<E, 128, 0, stream>>>(x_rel, alpha, eb, ej, ek, spk, agg, cnt, E);
  gemm_bf16_wmma<<<dim3(7, GY), 256, 0, stream>>>(feats_bf, 224, rootT, 224,
                                                  nullptr, tmp1, 100, N, 100, 224, 0);
  h1_combine<<<g1(N * 100), 256, 0, stream>>>(agg, cnt, tmp1, rbias, h1, N);
  pack_rows_bf16<<<g1(N * 128), 256, 0, stream>>>(h1, h1_bf, N, 100, 128);

  // --- GraphConv ---
  gc_msg<<<E, 128, 0, stream>>>(h1, eb, ej, ek, agg2, E);
  pack_rows_bf16<<<g1(N * 128), 256, 0, stream>>>(agg2, agg2_bf, N, 100, 128);
  gemm_bf16_wmma<<<dim3(7, GY), 256, 0, stream>>>(agg2_bf, 128, gcwT, 128,
                                                  nullptr, t1, 100, N, 100, 128, 0);
  gemm_bf16_wmma<<<dim3(7, GY), 256, 0, stream>>>(h1_bf, 128, gcrT, 128,
                                                  nullptr, t2, 100, N, 100, 128, 0);
  h2_combine<<<g1(N * 100), 256, 0, stream>>>(t1, t2, gcb, h2, N);

  // --- classifier ---
  pack_z<<<g1(N * 320), 256, 0, stream>>>(feats, h2, z_bf, N);
  gemm_bf16_wmma<<<dim3(7, GY), 256, 0, stream>>>(z_bf, 320, cw1T, 320, cb1, zz,
                                                  100, N, 100, 320, 1);
  clf_final<<<g1(N), 256, 0, stream>>>(zz, cw2, cb2, logits, N);
}
